// CRF_6648609374615
// MI455X (gfx1250) — matrix operand order repaired and checked by reading, not compiled, not run on hardware
//
#include <hip/hip_runtime.h>
#include <hip/hip_bf16.h>

#define SEQ_LEN    8192
#define TAGS       2048
#define MAIN_WGS   64
#define WG_THREADS 256
#define KCHUNKS    64          // 2048 / 32 total k-chunks per j-tile
#define KC_PER_WAVE 16         // K split 4 ways across waves (B slice fits in VGPRs)
#define JT_PER_WG  2           // 128 j-tiles / 64 WGs

typedef __attribute__((ext_vector_type(16))) __bf16 v16bf;
typedef __attribute__((ext_vector_type(8)))  float  v8f;

// float -> bf16, round-to-nearest-even
__device__ __forceinline__ unsigned short f2bf(float x) {
    unsigned u = __float_as_uint(x);
    u += 0x7FFFu + ((u >> 16) & 1u);
    return (unsigned short)(u >> 16);
}

// Offset inside a 32-wide K chunk matching the 16-bit A-matrix fragment:
// lane 0 holds K = 0..7,16..23 (v=0..15), lane 16 holds K = 8..15,24..31.
__device__ __forceinline__ int permA32(int r) {
    return r < 8 ? r : (r < 16 ? r + 8 : (r < 24 ? r - 8 : r));
}

// ---------------------------------------------------------------------------
// Prep: pack E = exp(transitions) into bf16 B-fragments, init fv0, reset the
// grid-barrier counter (must be re-zeroed every launch for graph replay).
// B 32x16 bf16 layout: lane L -> column N = L&15, K = (L<16?0:16)+v, v=0..15.
// ---------------------------------------------------------------------------
__global__ void __launch_bounds__(WG_THREADS)
crf_prep(const float* __restrict__ feats, const float* __restrict__ trans,
         const float* __restrict__ start, unsigned short* __restrict__ eb,
         float* __restrict__ fvbuf, unsigned* __restrict__ counter)
{
    size_t o = (size_t)blockIdx.x * WG_THREADS + threadIdx.x;  // < TAGS*TAGS
    int v    = (int)(o & 15);
    int lane = (int)((o >> 4) & 31);
    int kc   = (int)((o >> 9) & 63);
    int jt   = (int)(o >> 15);
    int j = jt * 16 + (lane & 15);
    int k = kc * 32 + (lane < 16 ? 0 : 16) + v;
    eb[o] = f2bf(__expf(trans[(size_t)j * TAGS + k]));
    if (o < TAGS) fvbuf[o] = start[o] + feats[o];   // fv0 into buffer 0
    if (o == 0)   *counter = 0u;
}

// ---------------------------------------------------------------------------
// Persistent cooperative forward recurrence: 64 WGs x 256 threads.
// Each WG owns 2 j-tiles; its 8 waves split K 4-ways per tile. Each wave's
// 16 B fragments (16 KB of E) are loop-invariant and live in VGPRs across
// all 8191 steps; per step only u (LDS) and fv/feats (L2) move.
// ---------------------------------------------------------------------------
__global__ void __launch_bounds__(WG_THREADS)
crf_forward(const float* __restrict__ feats, const unsigned short* __restrict__ eb,
            float* __restrict__ fvbuf, unsigned* __restrict__ counter)
{
    __shared__ float red[WG_THREADS];
    __shared__ __align__(32) unsigned short uperm[TAGS];  // bf16 u, A-fragment order
    __shared__ __align__(32) unsigned short zfrag[32];    // zero A rows 1..15
    __shared__ float part[JT_PER_WG][4][16];              // K-split partials

    const int tid   = threadIdx.x;
    const int lane  = tid & 31;
    const int wave  = tid >> 5;                       // 0..7
    const int jl    = wave >> 2;                      // local j-tile 0..1
    const int kq    = wave & 3;                       // K quadrant 0..3
    const int jt    = blockIdx.x * JT_PER_WG + jl;    // 0..127
    const bool arow0 = ((lane & 15) == 0);            // lanes 0 and 16 carry A row 0
    const int  ahalf = (lane >> 4) << 4;              // lane16 reads second 16 values

    if (tid < 32) zfrag[tid] = 0;
    __syncthreads();

    // This wave's slice of the B stream: j-tile jt, k-chunks kq*16 .. kq*16+15
    const unsigned short* ebw =
        eb + (size_t)(jt * KCHUNKS + kq * KC_PER_WAVE) * (32 * 16);
    // A fragment: base + stride (no per-iteration select)
    const unsigned short* abase =
        arow0 ? (uperm + kq * KC_PER_WAVE * 32 + ahalf) : zfrag;
    const int astride = arow0 ? 32 : 0;

    for (int t = 1; t < SEQ_LEN; ++t) {
        const float* fvin  = fvbuf + (((t - 1) & 1) << 11);
        float*       fvout = fvbuf + ((t & 1) << 11);

        // --- m = max(fv) (redundant per WG, 8 KB from L2) ---
        float lm = -3.4e38f;
        #pragma unroll
        for (int i = 0; i < TAGS / WG_THREADS; ++i)
            lm = fmaxf(lm, fvin[tid + i * WG_THREADS]);
        red[tid] = lm;
        __syncthreads();
        for (int s = WG_THREADS / 2; s > 0; s >>= 1) {
            if (tid < s) red[tid] = fmaxf(red[tid], red[tid + s]);
            __syncthreads();
        }
        const float m = red[0];

        // --- u = bf16(exp(fv - m)) stored in contiguous k order ---
        #pragma unroll
        for (int i = 0; i < TAGS / WG_THREADS; ++i) {
            int k = tid + i * WG_THREADS;
            uperm[k] = f2bf(__expf(fvin[k] - m));
        }
        __syncthreads();

        // --- partial w[j] over this wave's K slice: dual-accumulator WMMA ---
        // Two independent chains let the ds_load pairs overlap WMMA issue
        // instead of a strict load->wait->wmma serial chain.
        v8f acc0 = {}, acc1 = {};
        #pragma unroll
        for (int kc = 0; kc < KC_PER_WAVE; kc += 2) {
            v16bf a0 = *(const v16bf*)(abase + kc * astride);
            v16bf a1 = *(const v16bf*)(abase + (kc + 1) * astride);
            v16bf b0 = *(const v16bf*)(ebw + (size_t)((kc * 32 + lane) << 4));
            v16bf b1 = *(const v16bf*)(ebw + (size_t)(((kc + 1) * 32 + lane) << 4));
            acc0 = __builtin_amdgcn_wmma_f32_16x16x32_bf16(
                       false, a0, false, b0, (short)0, acc0, false, false);
            acc1 = __builtin_amdgcn_wmma_f32_16x16x32_bf16(
                       false, a1, false, b1, (short)0, acc1, false, false);
        }

        // D[0][N] sits in VGPR0 of lanes 0..15 (N = lane)
        if (lane < 16) part[jl][kq][lane] = acc0[0] + acc1[0];
        __syncthreads();

        // --- combine 4 K-partials, finish nfv, store (32 threads) ---
        if (tid < JT_PER_WG * 16) {
            int jj = tid >> 4, ln = tid & 15;
            float w = part[jj][0][ln] + part[jj][1][ln]
                    + part[jj][2][ln] + part[jj][3][ln];
            int j = (blockIdx.x * JT_PER_WG + jj) * 16 + ln;
            fvout[j] = m + __logf(w) + feats[(size_t)t * TAGS + j];
        }

        // --- grid-wide barrier (monotonic counter, target = 64*t) ---
        __threadfence();
        __syncthreads();
        if (tid == 0) {
            __hip_atomic_fetch_add(counter, 1u, __ATOMIC_RELEASE,
                                   __HIP_MEMORY_SCOPE_AGENT);
            const unsigned target = (unsigned)MAIN_WGS * (unsigned)t;
            while (__hip_atomic_load(counter, __ATOMIC_ACQUIRE,
                                     __HIP_MEMORY_SCOPE_AGENT) < target)
                __builtin_amdgcn_s_sleep(2);
        }
        __syncthreads();
    }
}

// ---------------------------------------------------------------------------
// Finalize: forward_score = logsumexp(fv_final + stop); gold path; out = diff.
// ---------------------------------------------------------------------------
__global__ void __launch_bounds__(256)
crf_final(const float* __restrict__ feats, const float* __restrict__ trans,
          const float* __restrict__ start, const float* __restrict__ stop,
          const int* __restrict__ tags, const float* __restrict__ fvf,
          float* __restrict__ out)
{
    __shared__ float red[256];
    const int tid = threadIdx.x;

    float lm = -3.4e38f;
    for (int i = tid; i < TAGS; i += 256) lm = fmaxf(lm, fvf[i] + stop[i]);
    red[tid] = lm; __syncthreads();
    for (int s = 128; s > 0; s >>= 1) {
        if (tid < s) red[tid] = fmaxf(red[tid], red[tid + s]);
        __syncthreads();
    }
    const float m = red[0];
    __syncthreads();

    float ls = 0.f;
    for (int i = tid; i < TAGS; i += 256) ls += __expf(fvf[i] + stop[i] - m);
    red[tid] = ls; __syncthreads();
    for (int s = 128; s > 0; s >>= 1) {
        if (tid < s) red[tid] += red[tid + s];
        __syncthreads();
    }
    const float forward = m + __logf(red[0]);
    __syncthreads();

    float g = 0.f;
    for (int t = tid + 1; t < SEQ_LEN; t += 256) {
        int cur = tags[t], prev = tags[t - 1];
        g += trans[(size_t)cur * TAGS + prev] + feats[(size_t)t * TAGS + cur];
    }
    if (tid == 0)
        g += start[tags[0]] + feats[tags[0]] + stop[tags[SEQ_LEN - 1]];
    red[tid] = g; __syncthreads();
    for (int s = 128; s > 0; s >>= 1) {
        if (tid < s) red[tid] += red[tid + s];
        __syncthreads();
    }
    if (tid == 0) out[0] = forward - red[0];
}

// ---------------------------------------------------------------------------
extern "C" void kernel_launch(void* const* d_in, const int* in_sizes, int n_in,
                              void* d_out, int out_size, void* d_ws, size_t ws_size,
                              hipStream_t stream)
{
    (void)in_sizes; (void)n_in; (void)out_size; (void)ws_size;
    const float* feats = (const float*)d_in[0];
    const float* trans = (const float*)d_in[1];
    const float* start = (const float*)d_in[2];
    const float* stop  = (const float*)d_in[3];
    const int*   tags  = (const int*)d_in[4];
    float* out = (float*)d_out;

    // workspace layout: [ E bf16 packed : 8 MB ][ fv ping-pong : 16 KB ][ counter ]
    char* ws = (char*)d_ws;
    unsigned short* eb      = (unsigned short*)ws;
    float*          fvbuf   = (float*)(ws + (size_t)TAGS * TAGS * 2);
    unsigned*       counter = (unsigned*)(ws + (size_t)TAGS * TAGS * 2
                                             + (size_t)2 * TAGS * 4);

    crf_prep<<<(TAGS * TAGS) / WG_THREADS, WG_THREADS, 0, stream>>>(
        feats, trans, start, eb, fvbuf, counter);
    crf_forward<<<MAIN_WGS, WG_THREADS, 0, stream>>>(feats, eb, fvbuf, counter);
    // final fv lives in buffer (8191 & 1) == 1
    crf_final<<<1, 256, 0, stream>>>(feats, trans, start, stop, tags,
                                     fvbuf + TAGS, out);
}
